// LegacySeqFirstSelfAttention_modified_LoRA_7679401525983
// MI455X (gfx1250) — matrix operand order repaired and checked by reading, not compiled, not run on hardware
//
#include <hip/hip_runtime.h>
#include <hip/hip_bf16.h>

// ---------------------------------------------------------------------------
// Sizes (fixed by the reference)
// ---------------------------------------------------------------------------
#define S_LEN   2048
#define BATCH   2
#define HIDDEN  1024
#define HEADS   16
#define D_HEAD  64
#define RANK    64
#define NROWS   (S_LEN * BATCH)        // 4096 rows (s-major, b-minor)
#define BH      (BATCH * HEADS)        // 32
#define CTX_ELEMS ((size_t)S_LEN * BATCH * HIDDEN)   // 4,194,304

typedef __attribute__((ext_vector_type(16))) __bf16 bf16x16;
typedef __attribute__((ext_vector_type(8)))  float  fx8;

union FragBF { bf16x16 v; uint4 u[2]; unsigned short s[16]; };
union FragF  { fx8 v; float f[8]; };

__device__ __forceinline__ unsigned short f2bf(float f) {
    unsigned int x = __float_as_uint(f);
    x += 0x7FFFu + ((x >> 16) & 1u);          // round-to-nearest-even
    return (unsigned short)(x >> 16);
}
__device__ __forceinline__ float bf2f(unsigned short u) {
    return __uint_as_float(((unsigned int)u) << 16);
}

__device__ __forceinline__ fx8 wmma_bf16(bf16x16 a, bf16x16 b, fx8 c) {
    // D = A(16x32) * B(32x16) + C, f32 accumulate
    return __builtin_amdgcn_wmma_f32_16x16x32_bf16(
        false, a, false, b, (short)0, c, false, false);
}

// A fragment: 16 rows x 32 k, row-major source. lane m=l&15 holds row m;
// elements 0..7 = row[k0 + h*8 .. +7], 8..15 = row[k0 + 16 + h*8 .. +7]
__device__ __forceinline__ void loadA(FragBF& fr, const unsigned short* base,
                                      int row0, int ldm, int k0, int lane) {
    int m = lane & 15, h = lane >> 4;
    const unsigned short* p = base + (size_t)(row0 + m) * ldm + k0 + h * 8;
    fr.u[0] = *reinterpret_cast<const uint4*>(p);
    fr.u[1] = *reinterpret_cast<const uint4*>(p + 16);
}

// B fragment: 32 k x 16 cols; source stores columns as contiguous rows
// (i.e. B^T row-major). lane n=l&15 holds column n; elements 0..7 =
// colrow[k0 + hb*8 .. +7], 8..15 = colrow[k0 + 16 + hb*8 .. +7]
__device__ __forceinline__ void loadB(FragBF& fr, const unsigned short* baseT,
                                      int col0, int ldm, int k0, int lane) {
    int n = lane & 15, hb = lane >> 4;
    const unsigned short* p = baseT + (size_t)(col0 + n) * ldm + k0 + hb * 8;
    fr.u[0] = *reinterpret_cast<const uint4*>(p);
    fr.u[1] = *reinterpret_cast<const uint4*>(p + 16);
}

__device__ __forceinline__ float sumsq16(const FragBF& fr) {
    float s = 0.f;
#pragma unroll
    for (int e = 0; e < 16; ++e) { float v = bf2f(fr.s[e]); s += v * v; }
    return s;
}

// ---------------------------------------------------------------------------
// Conversion kernels
// ---------------------------------------------------------------------------
__global__ void k_cvt(const float* __restrict__ in, unsigned short* __restrict__ out, int n) {
    int i = blockIdx.x * blockDim.x + threadIdx.x;
    if (i < n) out[i] = f2bf(in[i]);
}

// out[c][r] (bf16) = in[r][c] (f32);  in is rows x cols
__global__ void k_cvt_t(const float* __restrict__ in, unsigned short* __restrict__ out,
                        int rows, int cols) {
    int i = blockIdx.x * blockDim.x + threadIdx.x;
    if (i < rows * cols) {
        int r = i / cols, c = i % cols;
        out[(size_t)c * rows + r] = f2bf(in[i]);
    }
}

// ---------------------------------------------------------------------------
// Stage 1: tmp[mi] = x @ w{q,k,v}a + b{q,k,v}a      (M=4096, K=1024, N=64)
// grid (32, 3), block 128 (4 waves, each owns 32 rows)
// ---------------------------------------------------------------------------
__global__ __launch_bounds__(128)
void k_stage1(const unsigned short* __restrict__ Xbf,
              const unsigned short* __restrict__ waT,   // [3][64][1024] bf16
              const float* __restrict__ bq, const float* __restrict__ bk,
              const float* __restrict__ bv,
              unsigned short* __restrict__ tmp) {       // [3][4096][64] bf16
    const int mi   = blockIdx.y;
    const int lane = threadIdx.x & 31;
    const int wid  = threadIdx.x >> 5;
    const int mbase = blockIdx.x * 128 + wid * 32;
    const unsigned short* w = waT + (size_t)mi * 64 * 1024;
    const float* bias = (mi == 0) ? bq : (mi == 1) ? bk : bv;
    unsigned short* out = tmp + (size_t)mi * NROWS * RANK;

    FragF c[2][4];
#pragma unroll
    for (int st = 0; st < 2; ++st)
#pragma unroll
        for (int nt = 0; nt < 4; ++nt) c[st][nt].v = (fx8)0.f;

    for (int kb = 0; kb < HIDDEN; kb += 32) {
        FragBF a[2], bw[4];
#pragma unroll
        for (int st = 0; st < 2; ++st) loadA(a[st], Xbf, mbase + st * 16, HIDDEN, kb, lane);
#pragma unroll
        for (int nt = 0; nt < 4; ++nt) loadB(bw[nt], w, nt * 16, HIDDEN, kb, lane);
#pragma unroll
        for (int st = 0; st < 2; ++st)
#pragma unroll
            for (int nt = 0; nt < 4; ++nt)
                c[st][nt].v = wmma_bf16(a[st].v, bw[nt].v, c[st][nt].v);
    }

    const int h = lane >> 4, m = lane & 15;
#pragma unroll
    for (int st = 0; st < 2; ++st)
#pragma unroll
        for (int nt = 0; nt < 4; ++nt) {
            float bval = bias[nt * 16 + m];
#pragma unroll
            for (int r = 0; r < 8; ++r) {
                int row = mbase + st * 16 + r + 8 * h;
                out[(size_t)row * RANK + nt * 16 + m] = f2bf(c[st][nt].f[r] + bval);
            }
        }
}

// ---------------------------------------------------------------------------
// Stage 2: {Q,K,V} = tmp[mi] @ w{..}b + b{..}b      (M=4096, K=64, N=1024)
// Q,K stored [bh][s][64]; V stored transposed [bh][64][s].  grid (32,3)/128
// ---------------------------------------------------------------------------
__global__ __launch_bounds__(128)
void k_stage2(const unsigned short* __restrict__ tmp,
              const unsigned short* __restrict__ wbT,   // [3][1024][64] bf16
              const float* __restrict__ bq, const float* __restrict__ bk,
              const float* __restrict__ bv,
              unsigned short* __restrict__ Qbf,
              unsigned short* __restrict__ Kbf,
              unsigned short* __restrict__ Vt) {
    const int mi   = blockIdx.y;
    const int lane = threadIdx.x & 31;
    const int wid  = threadIdx.x >> 5;
    const int mbase = blockIdx.x * 128 + wid * 32;
    const unsigned short* in = tmp + (size_t)mi * NROWS * RANK;
    const unsigned short* w  = wbT + (size_t)mi * HIDDEN * RANK;
    const float* bias = (mi == 0) ? bq : (mi == 1) ? bk : bv;

    FragBF a[2][2];
#pragma unroll
    for (int st = 0; st < 2; ++st)
#pragma unroll
        for (int kc = 0; kc < 2; ++kc)
            loadA(a[st][kc], in, mbase + st * 16, RANK, kc * 32, lane);

    const int h = lane >> 4, m = lane & 15;
    for (int nt = 0; nt < HIDDEN / 16; ++nt) {
        const int n0 = nt * 16;
        FragBF bw[2];
#pragma unroll
        for (int kc = 0; kc < 2; ++kc) loadB(bw[kc], w, n0, RANK, kc * 32, lane);

        const int nloc = n0 + m;
        const float bval = bias[nloc];
        const int hh = nloc >> 6, dcol = nloc & 63;
#pragma unroll
        for (int st = 0; st < 2; ++st) {
            FragF c; c.v = (fx8)0.f;
            c.v = wmma_bf16(a[st][0].v, bw[0].v, c.v);
            c.v = wmma_bf16(a[st][1].v, bw[1].v, c.v);
#pragma unroll
            for (int r = 0; r < 8; ++r) {
                int row = mbase + st * 16 + r + 8 * h;
                int bb = row & 1, ss = row >> 1;
                int bh = bb * HEADS + hh;
                unsigned short val = f2bf(c.f[r] + bval);
                if (mi == 0)      Qbf[((size_t)bh * S_LEN + ss) * D_HEAD + dcol] = val;
                else if (mi == 1) Kbf[((size_t)bh * S_LEN + ss) * D_HEAD + dcol] = val;
                else              Vt [((size_t)bh * D_HEAD + dcol) * S_LEN + ss] = val;
            }
        }
    }
}

// ---------------------------------------------------------------------------
// Fused attention: per (bh, 128-s-tile) block. Each wave owns 32 s-rows.
// scores = 0.125*QK' - 0.0625*(|q|^2+|k|^2); write scores; ctx += exp(scores)@V
// grid (16, 32), block 128
// ---------------------------------------------------------------------------
__global__ __launch_bounds__(128)
void k_attn(const unsigned short* __restrict__ Qbf,
            const unsigned short* __restrict__ Kbf,
            const unsigned short* __restrict__ Vt,
            float* __restrict__ outCtx, float* __restrict__ outSc) {
    const int bh   = blockIdx.y;
    const int b    = bh >> 4;
    const int hh   = bh & 15;
    const int lane = threadIdx.x & 31;
    const int wid  = threadIdx.x >> 5;
    const int sbase = blockIdx.x * 128 + wid * 32;
    const int m = lane & 15, h = lane >> 4;

    __shared__ __align__(16) unsigned short pbuf[4][2][16][32];   // 8 KB, wave-private

    const unsigned short* Qb = Qbf + (size_t)bh * S_LEN * D_HEAD;
    const unsigned short* Kb = Kbf + (size_t)bh * S_LEN * D_HEAD;
    const unsigned short* Vb = Vt  + (size_t)bh * D_HEAD * S_LEN;
    float* scBase = outSc + (size_t)bh * S_LEN * S_LEN;

    // Q fragments (persistent) + per-row |q|^2
    FragBF qa[2][2];
    float qs[2][8];
#pragma unroll
    for (int st = 0; st < 2; ++st) {
        float loc = 0.f;
#pragma unroll
        for (int kc = 0; kc < 2; ++kc) {
            loadA(qa[st][kc], Qb, sbase + st * 16, D_HEAD, kc * 32, lane);
            loc += sumsq16(qa[st][kc]);
        }
        float tot = loc + __shfl_xor(loc, 16, 32);     // qsq[row m] in lanes m, m+16
#pragma unroll
        for (int r = 0; r < 8; ++r)
            qs[st][r] = __shfl(tot, r + 8 * h, 32);    // per-lane row broadcast
    }

    FragF ct[2][4];
#pragma unroll
    for (int st = 0; st < 2; ++st)
#pragma unroll
        for (int ds = 0; ds < 4; ++ds) ct[st][ds].v = (fx8)0.f;

    for (int tb = 0; tb < S_LEN; tb += 32) {
        // K fragments (B operand) + per-col |k|^2
        FragBF kf[2][2];
        float kq[2];
#pragma unroll
        for (int ts = 0; ts < 2; ++ts) {
            float loc = 0.f;
#pragma unroll
            for (int kc = 0; kc < 2; ++kc) {
                loadB(kf[ts][kc], Kb, tb + ts * 16, D_HEAD, kc * 32, lane);
                loc += sumsq16(kf[ts][kc]);
            }
            kq[ts] = loc + __shfl_xor(loc, 16, 32);    // ksq for col n = l&15
        }

        // scores tiles: write + exp -> LDS staging (D-layout -> A-layout)
#pragma unroll
        for (int st = 0; st < 2; ++st)
#pragma unroll
            for (int ts = 0; ts < 2; ++ts) {
                FragF d; d.v = (fx8)0.f;
                d.v = wmma_bf16(qa[st][0].v, kf[ts][0].v, d.v);
                d.v = wmma_bf16(qa[st][1].v, kf[ts][1].v, d.v);
#pragma unroll
                for (int r = 0; r < 8; ++r) {
                    float sc = 0.125f * d.f[r] - 0.0625f * (qs[st][r] + kq[ts]);
                    int srow = sbase + st * 16 + r + 8 * h;
                    int tcol = tb + ts * 16 + m;
                    scBase[(size_t)srow * S_LEN + tcol] = sc;
                    pbuf[wid][st][r + 8 * h][ts * 16 + m] = f2bf(__expf(sc));
                }
            }

        // P (A operand) from LDS; V (B operand) from transposed V
        FragBF pa[2], vf[4];
#pragma unroll
        for (int st = 0; st < 2; ++st) {
            const unsigned short* p = &pbuf[wid][st][m][h * 8];
            pa[st].u[0] = *reinterpret_cast<const uint4*>(p);
            pa[st].u[1] = *reinterpret_cast<const uint4*>(p + 16);
        }
#pragma unroll
        for (int ds = 0; ds < 4; ++ds) loadB(vf[ds], Vb, ds * 16, S_LEN, tb, lane);

#pragma unroll
        for (int st = 0; st < 2; ++st)
#pragma unroll
            for (int ds = 0; ds < 4; ++ds)
                ct[st][ds].v = wmma_bf16(pa[st].v, vf[ds].v, ct[st][ds].v);
    }

    // context: [S, B, HIDDEN]
#pragma unroll
    for (int st = 0; st < 2; ++st)
#pragma unroll
        for (int ds = 0; ds < 4; ++ds)
#pragma unroll
            for (int r = 0; r < 8; ++r) {
                int ss = sbase + st * 16 + r + 8 * h;
                int dcol = ds * 16 + m;
                outCtx[((size_t)ss * BATCH + b) * HIDDEN + hh * D_HEAD + dcol] =
                    ct[st][ds].f[r];
            }
}

// ---------------------------------------------------------------------------
// Host launcher
// ---------------------------------------------------------------------------
extern "C" void kernel_launch(void* const* d_in, const int* in_sizes, int n_in,
                              void* d_out, int out_size, void* d_ws, size_t ws_size,
                              hipStream_t stream) {
    const float* x   = (const float*)d_in[0];
    const float* wqa = (const float*)d_in[1];  const float* bqa = (const float*)d_in[2];
    const float* wqb = (const float*)d_in[3];  const float* bqb = (const float*)d_in[4];
    const float* wka = (const float*)d_in[5];  const float* bka = (const float*)d_in[6];
    const float* wkb = (const float*)d_in[7];  const float* bkb = (const float*)d_in[8];
    const float* wva = (const float*)d_in[9];  const float* bva = (const float*)d_in[10];
    const float* wvb = (const float*)d_in[11]; const float* bvb = (const float*)d_in[12];
    float* out = (float*)d_out;

    // Workspace carve-up (bytes)
    char* ws = (char*)d_ws;
    unsigned short* Xbf = (unsigned short*)(ws);                       //  8 MiB
    unsigned short* waT = (unsigned short*)(ws + 8388608);             //  384 KiB [3][64][1024]
    unsigned short* wbT = (unsigned short*)(ws + 8781824);             //  384 KiB [3][1024][64]
    unsigned short* tmp = (unsigned short*)(ws + 9175040);             // 1.5 MiB [3][4096][64]
    unsigned short* Qbf = (unsigned short*)(ws + 10747904);            //  8 MiB [32][2048][64]
    unsigned short* Kbf = (unsigned short*)(ws + 19136512);            //  8 MiB
    unsigned short* Vt  = (unsigned short*)(ws + 27525120);            //  8 MiB [32][64][2048]

    // 1) conversions
    k_cvt<<<(NROWS * HIDDEN + 255) / 256, 256, 0, stream>>>(x, Xbf, NROWS * HIDDEN);
    k_cvt_t<<<(HIDDEN * RANK + 255) / 256, 256, 0, stream>>>(wqa, waT,              HIDDEN, RANK);
    k_cvt_t<<<(HIDDEN * RANK + 255) / 256, 256, 0, stream>>>(wka, waT + 1 * 65536,  HIDDEN, RANK);
    k_cvt_t<<<(HIDDEN * RANK + 255) / 256, 256, 0, stream>>>(wva, waT + 2 * 65536,  HIDDEN, RANK);
    k_cvt_t<<<(RANK * HIDDEN + 255) / 256, 256, 0, stream>>>(wqb, wbT,              RANK, HIDDEN);
    k_cvt_t<<<(RANK * HIDDEN + 255) / 256, 256, 0, stream>>>(wkb, wbT + 1 * 65536,  RANK, HIDDEN);
    k_cvt_t<<<(RANK * HIDDEN + 255) / 256, 256, 0, stream>>>(wvb, wbT + 2 * 65536,  RANK, HIDDEN);

    // 2) LoRA stage 1: x @ Wa + ba
    k_stage1<<<dim3(NROWS / 128, 3), 128, 0, stream>>>(Xbf, waT, bqa, bka, bva, tmp);

    // 3) LoRA stage 2: tmp @ Wb + bb  -> Q, K (row-major), V (transposed)
    k_stage2<<<dim3(NROWS / 128, 3), 128, 0, stream>>>(tmp, wbT, bqb, bkb, bvb, Qbf, Kbf, Vt);

    // 4) fused distance-attention: scores out + exp + P@V
    k_attn<<<dim3(S_LEN / 128, BH), 128, 0, stream>>>(Qbf, Kbf, Vt, out, out + CTX_ELEMS);
}
